// CoordKANModel_46935402611182
// MI455X (gfx1250) — matrix operand order repaired and checked by reading, not compiled, not run on hardware
//
#include <hip/hip_runtime.h>
#include <hip/hip_bf16.h>
#include <math.h>

typedef _Float16 v16h __attribute__((ext_vector_type(16)));
typedef _Float16 v8h  __attribute__((ext_vector_type(8)));
typedef float    v8f  __attribute__((ext_vector_type(8)));

#define GRES   10
#define KORD   3
#define NB     (GRES + KORD)        // 13 basis funcs per input
#define NINTV  (GRES + 2 * KORD)    // 16 knot intervals
#define HID    32
#define KSPL   (HID * NB)           // 416 spline columns
#define KTOT   (KSPL + HID)         // 448 = spline + silu-base columns
#define KT_TILES (KTOT / 32)        // 14 WMMA K-tiles
#define WAVES  4
#define PTS_PER_WAVE 16
#define PTS_PER_WG   (WAVES * PTS_PER_WAVE)   // 64
#define THREADS (WAVES * 32)

__device__ __forceinline__ float siluf(float x) {
    return x / (1.0f + __expf(-x));
}

// Uniform cubic B-spline (knots = arange(-3,14)/G): only 4 basis funcs are
// nonzero at x; returns cell c (valid iff 0 <= c < 16) and weights for
// coefficient indices c-3 .. c.
__device__ __forceinline__ void bspline4(float x, int& c, float w[4]) {
    float t = x * (float)GRES + (float)KORD;       // x/h + K, h = 1/G
    t = fminf(fmaxf(t, -1.0f), 17.0f);
    float fc = floorf(t);
    c = (int)fc;
    float u  = t - fc;
    float u2 = u * u, u3 = u2 * u;
    float um = 1.0f - u;
    w[0] = um * um * um * (1.0f / 6.0f);
    w[1] = (3.0f * u3 - 6.0f * u2 + 4.0f) * (1.0f / 6.0f);
    w[2] = (-3.0f * u3 + 3.0f * u2 + 3.0f * u + 1.0f) * (1.0f / 6.0f);
    w[3] = u3 * (1.0f / 6.0f);
}

__global__ __launch_bounds__(THREADS)
void kan_fused_kernel(const float* __restrict__ coords,
                      const float* __restrict__ coef0, const float* __restrict__ sb0,
                      const float* __restrict__ sp0,
                      const float* __restrict__ coef1, const float* __restrict__ sb1,
                      const float* __restrict__ sp1,
                      const float* __restrict__ coef2, const float* __restrict__ sb2,
                      const float* __restrict__ sp2,
                      const float* __restrict__ dbias,
                      float* __restrict__ out, int nPts) {
    // ---- LDS staging (~105 KB / WG, fits 320 KB WGP pool) ----
    __shared__ _Float16 sWT[HID * KTOT];                 // layer1 weights^T, row n, col k (sp folded, sb appended)
    __shared__ _Float16 sA[WAVES][PTS_PER_WAVE * KTOT];  // per-wave A tiles (row-major, 16 x 448)
    __shared__ float sH0[PTS_PER_WG][HID];
    __shared__ float sH1[PTS_PER_WG][HID];
    __shared__ float sC0[2 * HID * NB];                  // layer0 coef (sp folded)
    __shared__ float sSb0[2 * HID];
    __shared__ float sC2[HID * NB];                      // layer2 coef (sp folded)
    __shared__ float sSb2[HID];

    const int tid  = threadIdx.x;
    const int lane = tid & 31;
    const int wave = tid >> 5;
    const int m    = lane & 15;   // row (M) / col (N) index inside fragment
    const int half = lane >> 4;   // which K-half of the fragment this lane holds
    const int wgBase = blockIdx.x * PTS_PER_WG;

    // ---- stage weights ----
    for (int idx = tid; idx < HID * KTOT; idx += THREADS) {
        int n = idx / KTOT, col = idx % KTOT;
        float v;
        if (col < KSPL) {
            int i = col / NB, k = col % NB;
            v = sp1[i * HID + n] * coef1[(i * HID + n) * NB + k];
        } else {
            v = sb1[(col - KSPL) * HID + n];
        }
        sWT[n * KTOT + col] = (_Float16)v;
    }
    for (int idx = tid; idx < 2 * HID * NB; idx += THREADS)
        sC0[idx] = sp0[idx / NB] * coef0[idx];
    for (int idx = tid; idx < 2 * HID; idx += THREADS) sSb0[idx] = sb0[idx];
    for (int idx = tid; idx < HID * NB; idx += THREADS)
        sC2[idx] = sp2[idx / NB] * coef2[idx];
    for (int idx = tid; idx < HID; idx += THREADS) sSb2[idx] = sb2[idx];
    __syncthreads();

    // ---- layer 0 (2 -> 32), tiny: plain VALU ----
    {
        int lp = wave * PTS_PER_WAVE + m;
        int gp = wgBase + lp;
        int gpc = gp < nPts ? gp : nPts - 1;
        float x0 = coords[2 * gpc + 0];
        float x1 = coords[2 * gpc + 1];
        int c0i, c1i; float wa[4], wb[4];
        bspline4(x0, c0i, wa);
        bspline4(x1, c1i, wb);
        bool in0 = (c0i >= 0) && (c0i < NINTV);
        bool in1 = (c1i >= 0) && (c1i < NINTV);
        float s0 = siluf(x0), s1 = siluf(x1);
        #pragma unroll
        for (int t = 0; t < 16; ++t) {
            int o = half * 16 + t;
            float acc = sSb0[o] * s0 + sSb0[HID + o] * s1;
            if (in0) {
                #pragma unroll
                for (int j = 0; j < 4; ++j) {
                    int k = c0i - 3 + j;
                    if (k >= 0 && k < NB) acc += wa[j] * sC0[o * NB + k];
                }
            }
            if (in1) {
                #pragma unroll
                for (int j = 0; j < 4; ++j) {
                    int k = c1i - 3 + j;
                    if (k >= 0 && k < NB) acc += wb[j] * sC0[(HID + o) * NB + k];
                }
            }
            sH0[lp][o] = acc;
        }
    }

    // ---- build layer-1 A tile: zero, then scatter 4 basis values + silu ----
    {
        _Float16* A = sA[wave];
        uint4* az = (uint4*)A;
        for (int q = lane; q < PTS_PER_WAVE * KTOT / 8; q += 32)
            az[q] = make_uint4(0u, 0u, 0u, 0u);
        for (int it = 0; it < PTS_PER_WAVE; ++it) {
            float h = sH0[wave * PTS_PER_WAVE + it][lane];
            _Float16* row = A + it * KTOT;
            row[KSPL + lane] = (_Float16)siluf(h);
            int c; float w[4];
            bspline4(h, c, w);
            if (c >= 0 && c < NINTV) {
                #pragma unroll
                for (int j = 0; j < 4; ++j) {
                    int k = c - 3 + j;
                    if (k >= 0 && k < NB) row[lane * NB + k] = (_Float16)w[j];
                }
            }
        }
    }

    // ---- layer 1 spline+base as 16x32 WMMA GEMM: A(16x448) x W^T(448x32) ----
    {
        const _Float16* A = sA[wave];
        v8f acc0 = {};
        v8f acc1 = {};
        #pragma unroll
        for (int kt = 0; kt < KT_TILES; ++kt) {
            int kb = kt * 32;
            // A fragment (16-bit 16x32): lane holds K = half*16 + {0..15}
            const _Float16* ar = A + m * KTOT + kb + half * 16;
            v8h alo = *(const v8h*)ar;
            v8h ahi = *(const v8h*)(ar + 8);
            v16h a;
            #pragma unroll
            for (int e = 0; e < 8; ++e) { a[e] = alo[e]; a[8 + e] = ahi[e]; }
            // B fragments (32x16): lane = col n, holds K = half*16 + {0..15}
            const _Float16* b0p = sWT + m * KTOT + kb + half * 16;
            const _Float16* b1p = sWT + (m + 16) * KTOT + kb + half * 16;
            v8h b0lo = *(const v8h*)b0p, b0hi = *(const v8h*)(b0p + 8);
            v8h b1lo = *(const v8h*)b1p, b1hi = *(const v8h*)(b1p + 8);
            v16h b0, b1;
            #pragma unroll
            for (int e = 0; e < 8; ++e) {
                b0[e] = b0lo[e]; b0[8 + e] = b0hi[e];
                b1[e] = b1lo[e]; b1[8 + e] = b1hi[e];
            }
            acc0 = __builtin_amdgcn_wmma_f32_16x16x32_f16(false, a, false, b0,
                                                          (short)0, acc0, false, false);
            acc1 = __builtin_amdgcn_wmma_f32_16x16x32_f16(false, a, false, b1,
                                                          (short)0, acc1, false, false);
        }
        // D layout: reg r -> M = r + 8*half, N = m
        #pragma unroll
        for (int r = 0; r < 8; ++r) {
            int lp = wave * PTS_PER_WAVE + r + half * 8;
            sH1[lp][m]      = acc0[r];
            sH1[lp][m + 16] = acc1[r];
        }
    }
    __syncthreads();

    // ---- layer 2 (32 -> 1) + sigmoid: split 16 inputs per half-wave ----
    {
        int lp = wave * PTS_PER_WAVE + m;
        float acc = 0.0f;
        #pragma unroll
        for (int ii = 0; ii < 16; ++ii) {
            int i = half * 16 + ii;
            float h = sH1[lp][i];
            acc += sSb2[i] * siluf(h);
            int c; float w[4];
            bspline4(h, c, w);
            if (c >= 0 && c < NINTV) {
                #pragma unroll
                for (int j = 0; j < 4; ++j) {
                    int k = c - 3 + j;
                    if (k >= 0 && k < NB) acc += w[j] * sC2[i * NB + k];
                }
            }
        }
        acc += __shfl_xor(acc, 16, 32);
        int gp = wgBase + lp;
        if (half == 0 && gp < nPts) {
            float z = acc + dbias[0];
            out[gp] = 1.0f / (1.0f + __expf(-z));
        }
    }
}

extern "C" void kernel_launch(void* const* d_in, const int* in_sizes, int n_in,
                              void* d_out, int out_size, void* d_ws, size_t ws_size,
                              hipStream_t stream) {
    // setup_inputs order: coords, grid0, coef0, sb0, sp0, grid1, coef1, sb1, sp1,
    //                     grid2, coef2, sb2, sp2, density_bias  (grids unused: uniform)
    const float* coords = (const float*)d_in[0];
    const float* coef0  = (const float*)d_in[2];
    const float* sb0    = (const float*)d_in[3];
    const float* sp0    = (const float*)d_in[4];
    const float* coef1  = (const float*)d_in[6];
    const float* sb1    = (const float*)d_in[7];
    const float* sp1    = (const float*)d_in[8];
    const float* coef2  = (const float*)d_in[10];
    const float* sb2    = (const float*)d_in[11];
    const float* sp2    = (const float*)d_in[12];
    const float* dbias  = (const float*)d_in[13];
    float* out = (float*)d_out;

    int nPts = in_sizes[0] / 2;                    // 262144
    int blocks = (nPts + PTS_PER_WG - 1) / PTS_PER_WG;  // 4096
    kan_fused_kernel<<<blocks, THREADS, 0, stream>>>(
        coords, coef0, sb0, sp0, coef1, sb1, sp1, coef2, sb2, sp2, dbias, out, nPts);
}
